// MyConv2d_49400713839096
// MI455X (gfx1250) — matrix operand order repaired and checked by reading, not compiled, not run on hardware
//
#include <hip/hip_runtime.h>

// MI455X / gfx1250: wave32, WMMA 16x16x32 f16 -> f32.
typedef __attribute__((ext_vector_type(16))) _Float16 v16h;
typedef __attribute__((ext_vector_type(8)))  _Float16 v8h;
typedef __attribute__((ext_vector_type(8)))  float    v8f;

#define C_IN   128
#define H_IMG  56
#define W_IMG  56
#define HW     3136            // 56*56, multiple of 16
#define O_CH   256
#define K_TOT  1152            // 9*128
#define BATCH  32
#define M_TOT  (BATCH*HW)      // 100352 positions, multiple of 128

// LDS row pitch in halves: 32 data + 8 pad -> 80B stride, conflict-free
// across 16 lanes on 64x4B banks (gcd(80,256)=16), still 16B-aligned.
#define APITCH 40
#define BPITCH 40

// ---------------------------------------------------------------------------
// Kernel 1: quantize p_c[256,128,3,3,7] -> wq[o][k] fp16, k = (kh*3+kw)*128+c.
// argmax over the 7 levels (positive norm scaling doesn't change argmax),
// strict-> keeps the first max, matching jnp.argmax.
// ---------------------------------------------------------------------------
__global__ __launch_bounds__(256)
void quant_weights(const float* __restrict__ p_c,
                   const float* __restrict__ q_level,
                   _Float16* __restrict__ wq) {
    int idx = blockIdx.x * 256 + threadIdx.x;     // over O*C_IN*9 = 294912
    if (idx >= O_CH * C_IN * 9) return;
    const float* p = p_c + (size_t)idx * 7;
    int best = 0; float bv = p[0];
    #pragma unroll
    for (int l = 1; l < 7; ++l) { float v = p[l]; if (v > bv) { bv = v; best = l; } }
    float qv = q_level[best];
    int e  = idx % 9;            // kh*3+kw
    int oi = idx / 9;
    int ci = oi % C_IN;
    int o  = oi / C_IN;
    wq[(size_t)o * K_TOT + e * C_IN + ci] = (_Float16)qv;
}

// ---------------------------------------------------------------------------
// Kernel 2: implicit-GEMM conv via V_WMMA_F32_16X16X32_F16.
//   C[O, P] = W[O, K] * X[K, P]
//   Block tile: 64 out-channels x 128 positions, 256 threads = 8 waves
//   arranged 4(M) x 2(N); each wave: 16 channels x 64 positions
//   = 4 accumulators -> 4 WMMAs per 32-wide K step, 36 K steps.
//   LDS: A tile [o][k] and B tile [p][k] (K-contiguous), padded rows so
//   fragment reads/stores are bank-conflict-free; every WMMA B fragment
//   is one aligned 32B LDS read.
// ---------------------------------------------------------------------------
__global__ __launch_bounds__(256)
void conv_wmma(const float* __restrict__ x,
               const _Float16* __restrict__ wq,
               float* __restrict__ out) {
    __shared__ __attribute__((aligned(32))) _Float16 lds_a[64 * APITCH];
    __shared__ __attribute__((aligned(32))) _Float16 lds_b[128 * BPITCH];

    const int tid  = threadIdx.x;
    const int lane = tid & 31;
    const int la   = lane & 15;
    const int hi   = lane >> 4;
    const int wave = tid >> 5;
    const int wm   = (wave & 3) * 16;     // wave's channel offset in block tile
    const int wn   = (wave >> 2) * 64;    // wave's position offset in block tile

    const int Pbase = blockIdx.x * 128;   // position tile
    const int Obase = blockIdx.y * 64;    // out-channel tile

    // ---- B staging invariants: thread owns (p_local, 16 channels)
    const int p_local = tid & 127;
    const int c_grp   = tid >> 7;                   // 0..1 (16 channels each)
    const int p_glob  = Pbase + p_local;
    const int n_img   = p_glob / HW;
    const int hw      = p_glob - n_img * HW;
    const int h       = hw / W_IMG;
    const int w       = hw - h * W_IMG;
    const float* xim  = x + (size_t)n_img * C_IN * HW;
    v16h* b_dst = (v16h*)&lds_b[p_local * BPITCH + c_grp * 16];

    // ---- A staging invariants: thread copies 8 contiguous k halves
    const int o_row = Obase + (tid >> 2);
    const int kc8   = (tid & 3) * 8;
    const _Float16* wrow = wq + (size_t)o_row * K_TOT + kc8;
    v8h* a_dst = (v8h*)&lds_a[(tid >> 2) * APITCH + kc8];

    v8f acc0 = {}; v8f acc1 = {}; v8f acc2 = {}; v8f acc3 = {};

    for (int t = 0; t < 36; ++t) {
        // stage A: 64x32 weight halves (16B coalesced copies)
        *a_dst = *(const v8h*)(wrow + t * 32);

        // stage B: 128 positions x 32 k (k = (kh*3+kw)*128 + c), K-contiguous
        const int r  = t >> 2;                 // 0..8
        const int kh = r / 3, kw = r - kh * 3;
        const int c0 = (t & 3) * 32;
        const int ih = h + kh - 1, iw = w + kw - 1;
        const bool valid = ((unsigned)ih < H_IMG) & ((unsigned)iw < W_IMG);
        const float* src = xim + (size_t)(c0 + c_grp * 16) * HW + ih * W_IMG + iw;
        v16h btmp;
        #pragma unroll
        for (int c = 0; c < 16; ++c) {
            float v = valid ? src[(size_t)c * HW] : 0.0f;
            btmp[c] = (_Float16)v;
        }
        *b_dst = btmp;                          // one 32B LDS store
        __syncthreads();

        // A fragment: lane = M row; halves 0..15 -> K hi*16+0..15
        v16h afrag;
        {
            const int row = wm + la;
            const v8h lo  = *(const v8h*)&lds_a[row * APITCH + hi * 16];
            const v8h hi8 = *(const v8h*)&lds_a[row * APITCH + hi * 16 + 8];
            #pragma unroll
            for (int i = 0; i < 8; ++i) { afrag[i] = lo[i]; afrag[i + 8] = hi8[i]; }
        }
        // B fragments: lane = N col; halves 0..15 = K hi*16+0..15
        // -> single aligned 32B read thanks to [p][k] layout
        const int colb = (wn + la) * BPITCH + hi * 16;
        v16h b0 = *(const v16h*)&lds_b[colb];
        v16h b1 = *(const v16h*)&lds_b[colb + 16 * BPITCH];
        v16h b2 = *(const v16h*)&lds_b[colb + 32 * BPITCH];
        v16h b3 = *(const v16h*)&lds_b[colb + 48 * BPITCH];

        acc0 = __builtin_amdgcn_wmma_f32_16x16x32_f16(false, afrag, false, b0,
                                                      (short)0, acc0, false, false);
        acc1 = __builtin_amdgcn_wmma_f32_16x16x32_f16(false, afrag, false, b1,
                                                      (short)0, acc1, false, false);
        acc2 = __builtin_amdgcn_wmma_f32_16x16x32_f16(false, afrag, false, b2,
                                                      (short)0, acc2, false, false);
        acc3 = __builtin_amdgcn_wmma_f32_16x16x32_f16(false, afrag, false, b3,
                                                      (short)0, acc3, false, false);
        __syncthreads();
    }

    // Epilogue: C col = lane&15 (position -> contiguous hw, coalesced stores),
    //           C row = hi*8 + reg (out channel).
    #pragma unroll
    for (int j = 0; j < 4; ++j) {
        const int pj  = Pbase + wn + j * 16;      // multiple of 16; HW%16==0
        const int ni  = pj / HW;
        const int hwb = pj - ni * HW;
        const v8f accv = (j == 0) ? acc0 : (j == 1) ? acc1 : (j == 2) ? acc2 : acc3;
        float* ob = out + (size_t)ni * O_CH * HW + hwb + la;
        const int orow0 = Obase + wm + hi * 8;
        #pragma unroll
        for (int rr = 0; rr < 8; ++rr) {
            ob[(size_t)(orow0 + rr) * HW] = accv[rr];
        }
    }
}

// ---------------------------------------------------------------------------
extern "C" void kernel_launch(void* const* d_in, const int* in_sizes, int n_in,
                              void* d_out, int out_size, void* d_ws, size_t ws_size,
                              hipStream_t stream) {
    const float* x       = (const float*)d_in[0];   // [32,128,56,56]
    const float* p_c     = (const float*)d_in[1];   // [256,128,3,3,7]
    const float* q_level = (const float*)d_in[2];   // [7]
    float* out = (float*)d_out;                     // [32,256,56,56]
    _Float16* wq = (_Float16*)d_ws;                 // [256][1152] fp16 = 576 KB

    (void)in_sizes; (void)n_in; (void)out_size; (void)ws_size;

    quant_weights<<<(O_CH * C_IN * 9 + 255) / 256, 256, 0, stream>>>(p_c, q_level, wq);

    dim3 grid(M_TOT / 128, O_CH / 64);              // 784 x 4
    conv_wmma<<<grid, 256, 0, stream>>>(x, wq, out);
}
